// SelfAttn_block_pos_batched_84619445666608
// MI455X (gfx1250) — matrix operand order repaired and checked by reading, not compiled, not run on hardware
//
#include <hip/hip_runtime.h>
#include <hip/hip_bf16.h>
#include <math.h>

// Problem constants
#define B_  16
#define L_  1024
#define C_  4
#define E_  256
#define H_  8
#define DH_ 32
#define BL_ (B_ * L_)          // 16384 rows
#define E3_ (3 * E_)           // 768
#define LN_EPS 1e-5f

typedef __attribute__((ext_vector_type(16))) __bf16 v16bf;
typedef __attribute__((ext_vector_type(8)))  float  v8f;

union FragBF {
    v16bf v;
    uint4 q[2];
    unsigned short u[16];
};

__device__ __forceinline__ unsigned short f2bf(float f) {
    unsigned int u = __float_as_uint(f);
    u += 0x7fffu + ((u >> 16) & 1u);       // round-to-nearest-even
    return (unsigned short)(u >> 16);
}

// ---- WMMA fragment loaders (bf16, 16x16x32, wave32 layouts per ISA 7.12.2) ----
// A (16x32, MxK), row-major src, leading dim ld (elements).
// lane = 16*hi + m ; holds K=[hi*8, hi*8+8) and K=[16+hi*8, 16+hi*8+8) of row m.
__device__ __forceinline__ v16bf load_a_bf16(const unsigned short* src, int ld, int lane) {
    int hi = lane >> 4, m = lane & 15;
    FragBF a;
    a.q[0] = *(const uint4*)(src + m * ld + hi * 8);
    a.q[1] = *(const uint4*)(src + m * ld + 16 + hi * 8);
    return a.v;
}

// B (32x16, KxN). Column n of B is contiguous in memory at srcT + n*ld (length >= 32).
// lane = 16*hi + n ; holds K=[hi*8, hi*8+8) and K=[16+hi*8, 16+hi*8+8) of column n.
__device__ __forceinline__ v16bf load_b_bf16(const unsigned short* srcT, int ld, int lane) {
    int hi = lane >> 4, n = lane & 15;
    FragBF b;
    const unsigned short* p = srcT + n * ld + hi * 8;
    b.q[0] = *(const uint4*)(p);
    b.q[1] = *(const uint4*)(p + 16);
    return b.v;
}

__device__ __forceinline__ v8f wmma_bf16(v16bf a, v16bf b, v8f c) {
    return __builtin_amdgcn_wmma_f32_16x16x32_bf16(false, a, false, b, (short)0, c, false, false);
}

// ---------------- block-wide sum over 256 threads (8 waves of 32) ----------------
__device__ __forceinline__ float block_sum256(float v, float* red) {
#pragma unroll
    for (int off = 16; off >= 1; off >>= 1) v += __shfl_xor(v, off, 32);
    int w = threadIdx.x >> 5;
    if ((threadIdx.x & 31) == 0) red[w] = v;
    __syncthreads();
    float t = red[0] + red[1] + red[2] + red[3] + red[4] + red[5] + red[6] + red[7];
    __syncthreads();
    return t;
}

// ================= Kernel 1: embedding gather + LayerNorm =================
__global__ __launch_bounds__(256) void k_embed_ln(
    const int* __restrict__ seq, const float* __restrict__ emb_w,
    const float* __restrict__ emb_b, float* __restrict__ emb_f32,
    unsigned short* __restrict__ emb_bf) {
    __shared__ float red[8];
    int row = blockIdx.x;                 // 0..BL-1
    int l = row & (L_ - 1);
    int e = threadIdx.x;
    int cls = seq[row];
    float x = emb_w[((size_t)l * E_ + e) * C_ + cls] + emb_b[l * E_ + e];
    float mean = block_sum256(x, red) * (1.0f / E_);
    float d = x - mean;
    float var = block_sum256(d * d, red) * (1.0f / E_);
    float y = d * rsqrtf(var + LN_EPS);
    emb_f32[(size_t)row * E_ + e] = y;
    emb_bf[(size_t)row * E_ + e]  = f2bf(y);
}

// ================= Kernel 2: cast weights to bf16 =================
__global__ void k_cast_weights(const float* __restrict__ wqkv_f,
                               const float* __restrict__ wout_f,
                               unsigned short* __restrict__ wqkv,
                               unsigned short* __restrict__ wout) {
    const int n1 = E3_ * E_;      // 196608
    const int n2 = E_ * E_;       // 65536
    for (int i = blockIdx.x * blockDim.x + threadIdx.x; i < n1 + n2;
         i += gridDim.x * blockDim.x) {
        if (i < n1) wqkv[i] = f2bf(wqkv_f[i]);
        else        wout[i - n1] = f2bf(wout_f[i - n1]);
    }
}

// ================= Kernel 3: QKV GEMM (bf16 WMMA, 16x64 per wave) ===============
// Y[n,m] = sum_k X[n,k] * W[m,k] + bias[m] ; X:[BL,E] bf16, W:[3E,E] bf16
// q,k -> [B,H,L,DH] bf16 ; v -> transposed vt [B,H,DH,L] bf16
__global__ __launch_bounds__(256) void k_qkv_gemm(
    const unsigned short* __restrict__ X, const unsigned short* __restrict__ W,
    const float* __restrict__ bias, unsigned short* __restrict__ qb,
    unsigned short* __restrict__ kb, unsigned short* __restrict__ vt) {
    int lane = threadIdx.x & 31;
    int gwave = blockIdx.x * 8 + (threadIdx.x >> 5);   // one 16x64 tile per wave
    const int tilesN = E3_ / 64;                        // 12 column groups
    int rowBase = (gwave / tilesN) * 16;
    int colBase = (gwave % tilesN) * 64;

    v8f acc[4] = {{}, {}, {}, {}};
#pragma unroll
    for (int kk = 0; kk < E_; kk += 32) {
        v16bf a = load_a_bf16(X + (size_t)rowBase * E_ + kk, E_, lane);   // reused 4x
#pragma unroll
        for (int cg = 0; cg < 4; cg++) {
            v16bf b = load_b_bf16(W + (size_t)(colBase + cg * 16) * E_ + kk, E_, lane);
            acc[cg] = wmma_bf16(a, b, acc[cg]);
        }
    }

    int hi = lane >> 4, n = lane & 15;
#pragma unroll
    for (int cg = 0; cg < 4; cg++) {
        int col = colBase + cg * 16 + n;
        float bv = bias[col];
        int which = col >> 8;            // 0=q 1=k 2=v
        int rem = col & (E_ - 1);
        int h = rem >> 5, dh = rem & (DH_ - 1);
#pragma unroll
        for (int r = 0; r < 8; r++) {
            int row = rowBase + r + 8 * hi;
            int b_ = row >> 10, l = row & (L_ - 1);
            unsigned short bf = f2bf(acc[cg][r] + bv);
            size_t bh = (size_t)b_ * H_ + h;
            if (which == 0)      qb[(bh * L_ + l) * DH_ + dh] = bf;
            else if (which == 1) kb[(bh * L_ + l) * DH_ + dh] = bf;
            else                 vt[(bh * DH_ + dh) * L_ + l] = bf;
        }
    }
}

// ================= Kernel 4: flash attention (64-key tiles per iteration) =======
__global__ __launch_bounds__(256) void k_flash_attn(
    const unsigned short* __restrict__ qb, const unsigned short* __restrict__ kb,
    const unsigned short* __restrict__ vt, unsigned short* __restrict__ ctx) {
    __shared__ unsigned short pshm[8][16 * 64];        // 2KB per wave
    int wave = threadIdx.x >> 5, lane = threadIdx.x & 31;
    int hi = lane >> 4, n = lane & 15;
    int gwave = blockIdx.x * 8 + wave;                 // B*H*(L/16) = 8192 waves
    int qtile = gwave & 63;
    int bh = gwave >> 6;
    int b_ = bh >> 3, h = bh & 7;

    const unsigned short* Q  = qb + (size_t)bh * L_ * DH_;
    const unsigned short* K  = kb + (size_t)bh * L_ * DH_;
    const unsigned short* VT = vt + (size_t)bh * DH_ * L_;
    unsigned short* myp = pshm[wave];

    const float scale = 0.1767766952966369f;           // 1/sqrt(32)

    v16bf a_q = load_a_bf16(Q + (size_t)qtile * 16 * DH_, DH_, lane);

    v8f acc0 = {}, acc1 = {};                          // ctx: dh 0..15 / 16..31
    float m_run[8], l_run[8];
#pragma unroll
    for (int r = 0; r < 8; r++) { m_run[r] = -1e30f; l_run[r] = 0.f; }

    for (int kt = 0; kt < L_; kt += 64) {
        // ---- scores: 4 S-tiles of 16 keys each (one WMMA each, K-dim = DH = 32)
        v8f z = {};
        v8f s[4];
#pragma unroll
        for (int t = 0; t < 4; t++) {
            v16bf bk = load_b_bf16(K + (size_t)(kt + 16 * t) * DH_, DH_, lane);
            s[t] = wmma_bf16(a_q, bk, z);
        }
        // ---- online softmax: one shuffle-reduction pass per row for 64 keys ----
        float p[4][8], corr[8];
#pragma unroll
        for (int r = 0; r < 8; r++) {
            float a0 = s[0][r] * scale, a1 = s[1][r] * scale;
            float a2 = s[2][r] * scale, a3 = s[3][r] * scale;
            float t = fmaxf(fmaxf(a0, a1), fmaxf(a2, a3));
#pragma unroll
            for (int off = 8; off >= 1; off >>= 1)
                t = fmaxf(t, __shfl_xor(t, off, 32));   // row-max over 16 lanes
            float newm = fmaxf(m_run[r], t);
            corr[r] = __expf(m_run[r] - newm);
            m_run[r] = newm;
            p[0][r] = __expf(a0 - newm);
            p[1][r] = __expf(a1 - newm);
            p[2][r] = __expf(a2 - newm);
            p[3][r] = __expf(a3 - newm);
            float sm = (p[0][r] + p[1][r]) + (p[2][r] + p[3][r]);
#pragma unroll
            for (int off = 8; off >= 1; off >>= 1)
                sm += __shfl_xor(sm, off, 32);          // row-sum over 16 lanes
            l_run[r] = l_run[r] * corr[r] + sm;
            acc0[r] *= corr[r];
            acc1[r] *= corr[r];
        }
        // ---- repack P (D-layout) -> A-layout via per-wave LDS (in-order, same wave)
#pragma unroll
        for (int r = 0; r < 8; r++) {
            int row = r + 8 * hi;
#pragma unroll
            for (int t = 0; t < 4; t++)
                myp[row * 64 + t * 16 + n] = f2bf(p[t][r]);
        }
        v16bf a_p0 = load_a_bf16(myp,      64, lane);   // keys kt   .. kt+32
        v16bf a_p1 = load_a_bf16(myp + 32, 64, lane);   // keys kt+32.. kt+64
        // ---- P @ V : Vt column n (= dh) contiguous over keys ----
        v16bf bv00 = load_b_bf16(VT + kt,                L_, lane);
        v16bf bv01 = load_b_bf16(VT + 16 * L_ + kt,      L_, lane);
        acc0 = wmma_bf16(a_p0, bv00, acc0);
        acc1 = wmma_bf16(a_p0, bv01, acc1);
        v16bf bv10 = load_b_bf16(VT + kt + 32,           L_, lane);
        v16bf bv11 = load_b_bf16(VT + 16 * L_ + kt + 32, L_, lane);
        acc0 = wmma_bf16(a_p1, bv10, acc0);
        acc1 = wmma_bf16(a_p1, bv11, acc1);
    }

    // ---- normalize + write ctx as bf16 [B,L,E] (E index = h*32 + dh) ----
#pragma unroll
    for (int r = 0; r < 8; r++) {
        int row = qtile * 16 + r + 8 * hi;
        float inv = 1.0f / l_run[r];
        size_t base = ((size_t)b_ * L_ + row) * E_ + h * DH_;
        ctx[base + n]      = f2bf(acc0[r] * inv);
        ctx[base + 16 + n] = f2bf(acc1[r] * inv);
    }
}

// ================= Kernel 5: output projection GEMM (16x64 per wave) ============
__global__ __launch_bounds__(256) void k_out_gemm(
    const unsigned short* __restrict__ X, const unsigned short* __restrict__ W,
    const float* __restrict__ bias, float* __restrict__ Y) {
    int lane = threadIdx.x & 31;
    int gwave = blockIdx.x * 8 + (threadIdx.x >> 5);
    const int tilesN = E_ / 64;                         // 4 column groups
    int rowBase = (gwave / tilesN) * 16;
    int colBase = (gwave % tilesN) * 64;

    v8f acc[4] = {{}, {}, {}, {}};
#pragma unroll
    for (int kk = 0; kk < E_; kk += 32) {
        v16bf a = load_a_bf16(X + (size_t)rowBase * E_ + kk, E_, lane);   // reused 4x
#pragma unroll
        for (int cg = 0; cg < 4; cg++) {
            v16bf b = load_b_bf16(W + (size_t)(colBase + cg * 16) * E_ + kk, E_, lane);
            acc[cg] = wmma_bf16(a, b, acc[cg]);
        }
    }
    int hi = lane >> 4, n = lane & 15;
#pragma unroll
    for (int cg = 0; cg < 4; cg++) {
        float bv = bias[colBase + cg * 16 + n];
#pragma unroll
        for (int r = 0; r < 8; r++) {
            int row = rowBase + r + 8 * hi;
            Y[(size_t)row * E_ + colBase + cg * 16 + n] = acc[cg][r] + bv;
        }
    }
}

// ================= Kernel 6: residual + final LayerNorm =================
__global__ __launch_bounds__(256) void k_final_ln(
    const float* __restrict__ attn, const float* __restrict__ emb,
    float* __restrict__ out) {
    __shared__ float red[8];
    size_t row = blockIdx.x;
    int e = threadIdx.x;
    float x = attn[row * E_ + e] + emb[row * E_ + e];
    float mean = block_sum256(x, red) * (1.0f / E_);
    float d = x - mean;
    float var = block_sum256(d * d, red) * (1.0f / E_);
    out[row * E_ + e] = d * rsqrtf(var + LN_EPS);
}

// ================= launch =================
extern "C" void kernel_launch(void* const* d_in, const int* in_sizes, int n_in,
                              void* d_out, int out_size, void* d_ws, size_t ws_size,
                              hipStream_t stream) {
    const int*   seq       = (const int*)d_in[0];
    const float* emb_w     = (const float*)d_in[1];
    const float* emb_b     = (const float*)d_in[2];
    const float* in_proj_w = (const float*)d_in[3];
    const float* in_proj_b = (const float*)d_in[4];
    const float* out_w     = (const float*)d_in[5];
    const float* out_b     = (const float*)d_in[6];
    float* out = (float*)d_out;

    // workspace carve-up (bytes, 256-aligned)
    char* ws = (char*)d_ws;
    size_t off = 0;
    auto take = [&](size_t bytes) { char* p = ws + off; off = (off + bytes + 255) & ~(size_t)255; return p; };
    float*          emb_f32 = (float*)take((size_t)BL_ * E_ * 4);          // 16 MB
    unsigned short* emb_bf  = (unsigned short*)take((size_t)BL_ * E_ * 2); //  8 MB
    unsigned short* wqkv    = (unsigned short*)take((size_t)E3_ * E_ * 2);
    unsigned short* wout    = (unsigned short*)take((size_t)E_ * E_ * 2);
    unsigned short* qb      = (unsigned short*)take((size_t)BL_ * E_ * 2);
    unsigned short* kb      = (unsigned short*)take((size_t)BL_ * E_ * 2);
    unsigned short* vt      = (unsigned short*)take((size_t)BL_ * E_ * 2);
    unsigned short* ctx     = (unsigned short*)take((size_t)BL_ * E_ * 2);
    float*          attn    = (float*)take((size_t)BL_ * E_ * 4);          // 16 MB

    k_embed_ln<<<BL_, 256, 0, stream>>>(seq, emb_w, emb_b, emb_f32, emb_bf);
    k_cast_weights<<<256, 256, 0, stream>>>(in_proj_w, out_w, wqkv, wout);
    // QKV GEMM: (16384/16)*(768/64) = 12288 wave-tiles / 8 waves = 1536 blocks
    k_qkv_gemm<<<1536, 256, 0, stream>>>(emb_bf, wqkv, in_proj_b, qb, kb, vt);
    // Flash attention: B*H*(L/16) = 8192 wave-tiles / 8 = 1024 blocks
    k_flash_attn<<<1024, 256, 0, stream>>>(qb, kb, vt, ctx);
    // Out GEMM: (16384/16)*(256/64) = 4096 wave-tiles / 8 = 512 blocks
    k_out_gemm<<<512, 256, 0, stream>>>(ctx, wout, out_b, attn);
    k_final_ln<<<BL_, 256, 0, stream>>>(attn, emb_f32, out);
    (void)in_sizes; (void)n_in; (void)out_size; (void)ws_size;
}
